// ResidualVectorQuantizer_27779848470536
// MI455X (gfx1250) — matrix operand order repaired and checked by reading, not compiled, not run on hardware
//
#include <hip/hip_runtime.h>

#define N_ROWS   32768
#define DIM      512
#define N_Q      8
#define BINS     2048

#define ROWS_PER_BLOCK 128
#define TPB 256
#define LDS_PITCH 520        // bf16 elements per bin row in LDS (512 + 8 pad)
#define BINS_PER_TILE 32
#define N_TILES (BINS / BINS_PER_TILE)

typedef __attribute__((ext_vector_type(16))) __bf16 v16bf;
typedef __attribute__((ext_vector_type(8)))  __bf16 v8bf;
typedef __attribute__((ext_vector_type(8)))  float  v8f;

static __device__ __forceinline__ unsigned short f32_to_bf16_rne(float x){
  unsigned int u = __float_as_uint(x);
  u += 0x7FFFu + ((u >> 16) & 1u);
  return (unsigned short)(u >> 16);
}
static __device__ __forceinline__ __bf16 bfc(float x){
  return __builtin_bit_cast(__bf16, f32_to_bf16_rne(x));
}

// CDNA5 async global->LDS copy (ASYNCcnt-tracked, no VGPR round trip).
// INST_OFFSET applies to BOTH the LDS and global address (ISA 08 S4.4),
// so one LDS-addr VGPR + one 32-bit global offset VGPR + SGPR base cover
// a 128B chunk with 8 b128 copies varying only the immediate.
#define ASYNC_B128(LO, VO, GB, OFF) \
  asm volatile("global_load_async_to_lds_b128 %0, %1, %2 offset:" OFF \
               :: "v"(LO), "v"(VO), "s"(GB) : "memory")
#define WAIT_ASYNC() asm volatile("s_wait_asynccnt 0x0" ::: "memory")

// ---------------------------------------------------------------- prep kernels
__global__ void rvq_init_residual(const float4* __restrict__ src,
                                  float4* __restrict__ dst, int n4){
  int i = blockIdx.x * blockDim.x + threadIdx.x;
  int stride = gridDim.x * blockDim.x;
  for (; i < n4; i += stride) dst[i] = src[i];
}

// one block per (stage, bin): convert 512 f32 -> bf16 and compute ||c||^2 in f32
__global__ void rvq_prep_codebook(const float* __restrict__ cb,
                                  unsigned short* __restrict__ cbbf,
                                  float* __restrict__ csq){
  const int b = blockIdx.x;                 // q*BINS + bin
  const float* src = cb + (size_t)b * DIM;
  unsigned short* dst = cbbf + (size_t)b * DIM;
  float s = 0.f;
  for (int i = threadIdx.x; i < DIM; i += TPB){
    float x = src[i];
    s += x * x;
    dst[i] = f32_to_bf16_rne(x);
  }
  __shared__ float red[TPB];
  red[threadIdx.x] = s;
  __syncthreads();
  for (int o = TPB / 2; o > 0; o >>= 1){
    if (threadIdx.x < o) red[threadIdx.x] += red[threadIdx.x + o];
    __syncthreads();
  }
  if (threadIdx.x == 0) csq[b] = red[0];
}

// ---------------------------------------------------------------- stage kernel
// One block = 128 rows (8 waves x 16 rows). Sweep 2048 bins in 32-bin tiles
// (two independent WMMA accumulator chains per tile to hide XDL latency).
// B tiles are double-buffered in LDS and filled with async global->LDS copies.
// Score = ||c||^2 - 2 r.c  (||r||^2 dropped: constant per row, argmin-invariant).
__global__ __launch_bounds__(TPB) void rvq_stage(
    const float* __restrict__ cb_f32,         // [BINS, DIM] this stage (f32, exact update)
    const unsigned short* __restrict__ cb_bf, // [BINS, DIM] this stage (bf16, WMMA)
    const float* __restrict__ csq,            // [BINS]     this stage
    float* __restrict__ residual,             // [N, DIM]   f32, updated in place
    float* __restrict__ codes_out)            // [N]        this stage (argmin as float)
{
  __shared__ __align__(16) unsigned short ldsB[2][BINS_PER_TILE * LDS_PITCH];
  __shared__ int ldsIdx[ROWS_PER_BLOCK];

  const int tid  = threadIdx.x;
  const int wave = tid >> 5;
  const int lane = tid & 31;
  const int m    = lane & 15;   // row (A) / column (B,D) selector within fragment
  const int g    = lane >> 4;   // lane-group per ISA 7.12.2 layouts
  const int rowbase = blockIdx.x * ROWS_PER_BLOCK + wave * 16;

  // ---- load + convert this wave's 16x512 A tile into 16 bf16 WMMA fragments.
  // 16-bit A 16x32 layout: elem e<8 -> K = k0 + 8g + e ; e>=8 -> K = k0 + 16 + 8g + (e-8)
  v16bf A[16];
  const float* rrow = residual + (size_t)(rowbase + m) * DIM;
  #pragma unroll
  for (int f = 0; f < 16; ++f){
    const int k0 = f * 32;
    const float4 x0 = *(const float4*)(rrow + k0 + 8 * g);
    const float4 x1 = *(const float4*)(rrow + k0 + 8 * g + 4);
    const float4 y0 = *(const float4*)(rrow + k0 + 16 + 8 * g);
    const float4 y1 = *(const float4*)(rrow + k0 + 16 + 8 * g + 4);
    v16bf a;
    a[0]  = bfc(x0.x); a[1]  = bfc(x0.y); a[2]  = bfc(x0.z); a[3]  = bfc(x0.w);
    a[4]  = bfc(x1.x); a[5]  = bfc(x1.y); a[6]  = bfc(x1.z); a[7]  = bfc(x1.w);
    a[8]  = bfc(y0.x); a[9]  = bfc(y0.y); a[10] = bfc(y0.z); a[11] = bfc(y0.w);
    a[12] = bfc(y1.x); a[13] = bfc(y1.y); a[14] = bfc(y1.z); a[15] = bfc(y1.w);
    A[f] = a;
  }

  float best[8];
  int   bidx[8];
  #pragma unroll
  for (int e = 0; e < 8; ++e){ best[e] = 3.4e38f; bidx[e] = 0; }

  // async tile prefetch: 256 threads x 128B = 32 bins x 1KB (padded pitch in LDS)
  const unsigned fill_bin  = (unsigned)(tid >> 3);   // 0..31
  const unsigned fill_part = (unsigned)(tid & 7);    // 128B chunk within bin row
  const unsigned fill_lds_off = fill_bin * (LDS_PITCH * 2) + fill_part * 128;
  const unsigned fill_g_off   = fill_bin * (DIM * 2)       + fill_part * 128;

  auto issue_tile = [&](int tt){
    const unsigned lo = (unsigned)(size_t)(&ldsB[tt & 1][0]) + fill_lds_off;
    const unsigned long long gb =
        (unsigned long long)(size_t)cb_bf +
        (unsigned long long)tt * (unsigned long long)(BINS_PER_TILE * DIM * 2);
    const unsigned vo = fill_g_off;
    ASYNC_B128(lo, vo, gb, "0");   ASYNC_B128(lo, vo, gb, "16");
    ASYNC_B128(lo, vo, gb, "32");  ASYNC_B128(lo, vo, gb, "48");
    ASYNC_B128(lo, vo, gb, "64");  ASYNC_B128(lo, vo, gb, "80");
    ASYNC_B128(lo, vo, gb, "96");  ASYNC_B128(lo, vo, gb, "112");
  };

  issue_tile(0);

  for (int t = 0; t < N_TILES; ++t){
    WAIT_ASYNC();          // my async writes for tile t have landed in LDS
    __syncthreads();       // everyone's writes visible; everyone done with t-1
    if (t + 1 < N_TILES) issue_tile(t + 1);   // prefetch into the other buffer

    const unsigned short* buf = &ldsB[t & 1][0];
    const int binbase = t * BINS_PER_TILE;

    v8f acc0 = {0.f,0.f,0.f,0.f,0.f,0.f,0.f,0.f};
    v8f acc1 = {0.f,0.f,0.f,0.f,0.f,0.f,0.f,0.f};
    const unsigned short* lb0 = buf + m * LDS_PITCH;          // columns binbase+m
    const unsigned short* lb1 = buf + (16 + m) * LDS_PITCH;   // columns binbase+16+m
    #pragma unroll
    for (int f = 0; f < 16; ++f){
      // 16-bit B 32x16 layout: elem e<8 -> K = k0 + 8g + e ; e>=8 -> K = k0 + 16 + 8g + (e-8)
      const v8bf a0 = *(const v8bf*)(lb0 + f * 32 + 8 * g);
      const v8bf a1 = *(const v8bf*)(lb0 + f * 32 + 16 + 8 * g);
      const v8bf c0 = *(const v8bf*)(lb1 + f * 32 + 8 * g);
      const v8bf c1 = *(const v8bf*)(lb1 + f * 32 + 16 + 8 * g);
      const v16bf B0 = __builtin_shufflevector(a0, a1,
          0,1,2,3,4,5,6,7,8,9,10,11,12,13,14,15);
      const v16bf B1 = __builtin_shufflevector(c0, c1,
          0,1,2,3,4,5,6,7,8,9,10,11,12,13,14,15);
      acc0 = __builtin_amdgcn_wmma_f32_16x16x32_bf16(
          false, A[f], false, B0, (short)0, acc0, false, false);
      acc1 = __builtin_amdgcn_wmma_f32_16x16x32_bf16(
          false, A[f], false, B1, (short)0, acc1, false, false);
    }

    const float cs0 = csq[binbase + m];
    const float cs1 = csq[binbase + 16 + m];
    #pragma unroll
    for (int e = 0; e < 8; ++e){          // D layout: row = e + 8g, col = m
      const float d0 = fmaf(-2.f, acc0[e], cs0);
      if (d0 < best[e]){ best[e] = d0; bidx[e] = binbase + m; }
      const float d1 = fmaf(-2.f, acc1[e], cs1);
      if (d1 < best[e]){ best[e] = d1; bidx[e] = binbase + 16 + m; }
    }
  }

  // ---- argmin reduction across the 16 columns of each row (xor-shuffle within group)
  #pragma unroll
  for (int e = 0; e < 8; ++e){
    float b = best[e]; int bi = bidx[e];
    #pragma unroll
    for (int off = 8; off >= 1; off >>= 1){
      const float ob = __shfl_xor(b, off, 32);
      const int   oi = __shfl_xor(bi, off, 32);
      if (ob < b || (ob == b && oi < bi)){ b = ob; bi = oi; }  // first-index tiebreak
    }
    best[e] = b; bidx[e] = bi;
  }
  if (m == 0){
    #pragma unroll
    for (int e = 0; e < 8; ++e){
      const int row = e + 8 * g;
      codes_out[rowbase + row] = (float)bidx[e];
      ldsIdx[wave * 16 + row] = bidx[e];
    }
  }
  __syncthreads();

  // ---- residual -= chosen codeword, using the ORIGINAL f32 codebook (no bf16 drift)
  for (int r = 0; r < 16; ++r){
    const int idx = ldsIdx[wave * 16 + r];
    const float4* c = (const float4*)(cb_f32 + (size_t)idx * DIM) + lane * 4;
    float4* rp = (float4*)(residual + (size_t)(rowbase + r) * DIM) + lane * 4;
    #pragma unroll
    for (int j = 0; j < 4; ++j){
      float4 rv = rp[j];
      const float4 cv = c[j];
      rv.x -= cv.x; rv.y -= cv.y; rv.z -= cv.z; rv.w -= cv.w;
      rp[j] = rv;
    }
  }
}

// ---------------------------------------------------------------- finalize
// quantized = hidden - residual_final  (== sum of chosen codewords, exactly)
__global__ void rvq_finalize(const float4* __restrict__ hidden,
                             const float4* __restrict__ residual,
                             float4* __restrict__ quant, int n4){
  int i = blockIdx.x * blockDim.x + threadIdx.x;
  int stride = gridDim.x * blockDim.x;
  for (; i < n4; i += stride){
    float4 h = hidden[i], r = residual[i];
    float4 q; q.x = h.x - r.x; q.y = h.y - r.y; q.z = h.z - r.z; q.w = h.w - r.w;
    quant[i] = q;
  }
}

// ---------------------------------------------------------------- launch
extern "C" void kernel_launch(void* const* d_in, const int* in_sizes, int n_in,
                              void* d_out, int out_size, void* d_ws, size_t ws_size,
                              hipStream_t stream){
  (void)in_sizes; (void)n_in; (void)out_size; (void)ws_size;

  const float* hidden = (const float*)d_in[0];
  const float* cb     = (const float*)d_in[1];

  float* out   = (float*)d_out;
  float* codes = out;                               // [N_Q, N] (indices as float)
  float* quant = out + (size_t)N_Q * N_ROWS;        // [N, DIM]

  // workspace layout: residual f32 (64MB) | cb bf16 (16MB) | csq f32 (64KB)
  char* ws = (char*)d_ws;
  float* residual       = (float*)ws;
  unsigned short* cbbf  = (unsigned short*)(ws + (size_t)N_ROWS * DIM * sizeof(float));
  float* csq            = (float*)(ws + (size_t)N_ROWS * DIM * sizeof(float)
                                      + (size_t)N_Q * BINS * DIM * sizeof(unsigned short));

  const int n4 = N_ROWS * DIM / 4;
  rvq_init_residual<<<2048, TPB, 0, stream>>>((const float4*)hidden, (float4*)residual, n4);
  rvq_prep_codebook<<<N_Q * BINS, TPB, 0, stream>>>(cb, cbbf, csq);

  for (int s = 0; s < N_Q; ++s){
    rvq_stage<<<N_ROWS / ROWS_PER_BLOCK, TPB, 0, stream>>>(
        cb   + (size_t)s * BINS * DIM,
        cbbf + (size_t)s * BINS * DIM,
        csq  + (size_t)s * BINS,
        residual,
        codes + (size_t)s * N_ROWS);
  }

  rvq_finalize<<<2048, TPB, 0, stream>>>((const float4*)hidden, (const float4*)residual,
                                         (float4*)quant, n4);
}
